// CrossAttention_5385888989393
// MI455X (gfx1250) — matrix operand order repaired and checked by reading, not compiled, not run on hardware
//
#include <hip/hip_runtime.h>
#include <hip/hip_bf16.h>

typedef __attribute__((ext_vector_type(16))) __bf16 v16bf;
typedef __attribute__((ext_vector_type(8)))  __bf16 v8bf;
typedef __attribute__((ext_vector_type(8)))  float  v8f;
typedef __attribute__((ext_vector_type(4)))  float  v4f;

__device__ __forceinline__ v16bf cat8(v8bf lo, v8bf hi) {
  return __builtin_shufflevector(lo, hi, 0,1,2,3,4,5,6,7,8,9,10,11,12,13,14,15);
}

// Generic C = A * B^T  (A: MxK row-major contiguous-K, B: NxK row-major contiguous-K)
// B_KN: B source is actually KxN (contiguous N) f32 -> transposed while staged into LDS.
// Per-z offsets: z -> (bb = z/ZH, hh = z%ZH); base += bb*s?b + hh*s?h (element units).
template<int BM, int BN, int BK, int WM, int WN,
         bool A_F32, bool B_F32, bool B_KN,
         bool OUT_BF16, bool TRANS_C, bool HAS_BIAS>
__global__ __launch_bounds__((BM/WM)*(BN/WN)*32)
void wmma_gemm_nt(const void* __restrict__ Ag, int lda, long long sAb, long long sAh,
                  const void* __restrict__ Bg, int ldb, long long sBb, long long sBh,
                  void* __restrict__ Cg, int ldc, long long sCb, long long sCh,
                  int K, int ZH, float scale, const float* __restrict__ bias)
{
  constexpr int WAVES = (BM/WM)*(BN/WN);
  constexpr int NT    = WAVES*32;
  constexpr int LDSS  = BK + 8;          // bf16 units; 72*2B = 144B rows, 16B aligned
  constexpr int TM    = WM/16, TN = WN/16;
  constexpr int AV    = A_F32 ? 4 : 8;
  constexpr int BV    = B_F32 ? 4 : 8;

  __shared__ __bf16 As[BM*LDSS];
  __shared__ __bf16 Bs[BN*LDSS];

  const int z  = blockIdx.z;
  const int bb = z / ZH, hh = z % ZH;
  const int tid  = threadIdx.x;
  const int lane = tid & 31;
  const int w    = tid >> 5;
  const int wm   = w % (BM/WM);
  const int wn   = w / (BM/WM);
  const int rowbase = blockIdx.x * BM;
  const int colbase = blockIdx.y * BN;

  const float*  Af32 = (const float*) Ag + (long long)bb*sAb + (long long)hh*sAh;
  const __bf16* Abf  = (const __bf16*)Ag + (long long)bb*sAb + (long long)hh*sAh;
  const float*  Bf32 = (const float*) Bg + (long long)bb*sBb + (long long)hh*sBh;
  const __bf16* Bbf  = (const __bf16*)Bg + (long long)bb*sBb + (long long)hh*sBh;

  v8f acc[TM][TN];
  #pragma unroll
  for (int i = 0; i < TM; ++i)
    #pragma unroll
    for (int j = 0; j < TN; ++j)
      #pragma unroll
      for (int r = 0; r < 8; ++r) acc[i][j][r] = 0.0f;

  for (int k0 = 0; k0 < K; k0 += BK) {
    // ---- stage A tile (BM x BK) into LDS as bf16, coalesced along K ----
    #pragma unroll
    for (int p = 0; p < (BM*BK)/(NT*AV); ++p) {
      int flat = (p*NT + tid) * AV;
      int r = flat / BK, c = flat % BK;
      if (A_F32) {
        v4f v = *(const v4f*)(Af32 + (long long)(rowbase + r)*lda + (k0 + c));
        #pragma unroll
        for (int i = 0; i < 4; ++i) As[r*LDSS + c + i] = (__bf16)v[i];
      } else {
        v8bf v = *(const v8bf*)(Abf + (long long)(rowbase + r)*lda + (k0 + c));
        *(v8bf*)&As[r*LDSS + c] = v;
      }
    }
    // ---- stage B tile ----
    if (B_KN) {
      // source is KxN f32 (contiguous N): read coalesced along N, store transposed
      #pragma unroll
      for (int p = 0; p < (BK*BN)/(NT*4); ++p) {
        int flat = (p*NT + tid) * 4;
        int r = flat / BN, c = flat % BN;          // r = k, c = n
        v4f v = *(const v4f*)(Bf32 + (long long)(k0 + r)*ldb + (colbase + c));
        #pragma unroll
        for (int i = 0; i < 4; ++i) Bs[(c + i)*LDSS + r] = (__bf16)v[i];
      }
    } else {
      #pragma unroll
      for (int p = 0; p < (BN*BK)/(NT*BV); ++p) {
        int flat = (p*NT + tid) * BV;
        int r = flat / BK, c = flat % BK;
        if (B_F32) {
          v4f v = *(const v4f*)(Bf32 + (long long)(colbase + r)*ldb + (k0 + c));
          #pragma unroll
          for (int i = 0; i < 4; ++i) Bs[r*LDSS + c + i] = (__bf16)v[i];
        } else {
          v8bf v = *(const v8bf*)(Bbf + (long long)(colbase + r)*ldb + (k0 + c));
          *(v8bf*)&Bs[r*LDSS + c] = v;
        }
      }
    }
    __syncthreads();

    // ---- WMMA over this K-slab ----
    #pragma unroll
    for (int kk = 0; kk < BK; kk += 32) {
      // A frag (16x32 bf16): lanes 0-15 rows M, K chunks {0..7,16..23}; lanes 16-31 {8..15,24..31}
      const int akb = kk + ((lane >> 4) << 3);
      // B frag (32x16 bf16): lane = column N, lanes 0-15 K chunks {0..7,16..23}; lanes 16-31 {8..15,24..31}
      const int bkb = kk + ((lane >> 4) << 3);
      const int am  = wm*WM + (lane & 15);
      const int bn  = wn*WN + (lane & 15);
      v16bf af[TM], bfr[TN];
      #pragma unroll
      for (int i = 0; i < TM; ++i) {
        const __bf16* pa = &As[(am + i*16)*LDSS + akb];
        af[i] = cat8(*(const v8bf*)pa, *(const v8bf*)(pa + 16));
      }
      #pragma unroll
      for (int j = 0; j < TN; ++j) {
        const __bf16* pb = &Bs[(bn + j*16)*LDSS + bkb];
        bfr[j] = cat8(*(const v8bf*)pb, *(const v8bf*)(pb + 16));
      }
      #pragma unroll
      for (int i = 0; i < TM; ++i)
        #pragma unroll
        for (int j = 0; j < TN; ++j)
          acc[i][j] = __builtin_amdgcn_wmma_f32_16x16x32_bf16(
              false, af[i], false, bfr[j], (short)0, acc[i][j], false, false);
    }
    __syncthreads();
  }

  // ---- epilogue: C 16x16 f32 layout: VGPR r -> M = r + 8*(lane>=16), N = lane&15 ----
  float*  Cf = (float*) Cg + (long long)bb*sCb + (long long)hh*sCh;
  __bf16* Cb = (__bf16*)Cg + (long long)bb*sCb + (long long)hh*sCh;
  const int rhi = (lane >> 4) << 3;
  const int cl  = lane & 15;
  #pragma unroll
  for (int i = 0; i < TM; ++i)
    #pragma unroll
    for (int j = 0; j < TN; ++j)
      #pragma unroll
      for (int r = 0; r < 8; ++r) {
        int gm = rowbase + wm*WM + i*16 + r + rhi;
        int gn = colbase + wn*WN + j*16 + cl;
        float v = acc[i][j][r] * scale;
        if (HAS_BIAS) v += bias[gm];
        long long idx = TRANS_C ? ((long long)gn*ldc + gm) : ((long long)gm*ldc + gn);
        if (OUT_BF16) Cb[idx] = (__bf16)v; else Cf[idx] = v;
      }
}

// Row softmax of 64x64 dots, written transposed as bf16 (so next GEMM is NT).
__global__ void softmax_transpose_64(const float* __restrict__ dots,
                                     __bf16* __restrict__ attnT) {
  const int z = blockIdx.x;        // (b*8+h)
  const int i = threadIdx.x;       // row 0..63
  const float* row = dots + (long long)z*4096 + i*64;
  float m = row[0];
  #pragma unroll
  for (int j = 1; j < 64; ++j) m = fmaxf(m, row[j]);
  float e[64], s = 0.f;
  #pragma unroll
  for (int j = 0; j < 64; ++j) { e[j] = __expf(row[j] - m); s += e[j]; }
  float inv = 1.0f / s;
  __bf16* o = attnT + (long long)z*4096 + i;
  #pragma unroll
  for (int j = 0; j < 64; ++j) o[j*64] = (__bf16)(e[j]*inv);
}

// WvT[c][p] = Wkv[512+p][c] as bf16 (v-half of Wkv, transposed for NT GEMM).
__global__ void transpose_wv_bf16(const float* __restrict__ Wkv,
                                  __bf16* __restrict__ WvT) {
  int idx = blockIdx.x*256 + threadIdx.x;   // 512*512
  int c = idx >> 9, p = idx & 511;
  WvT[idx] = (__bf16)Wkv[(512 + p)*512 + c];
}

extern "C" void kernel_launch(void* const* d_in, const int* in_sizes, int n_in,
                              void* d_out, int out_size, void* d_ws, size_t ws_size,
                              hipStream_t stream) {
  const float* f_m  = (const float*)d_in[0];   // 8 x 512 x 4096
  const float* f_n  = (const float*)d_in[1];   // 8 x 512 x 4096
  const float* Wq   = (const float*)d_in[2];   // 512 x 512
  const float* Wkv  = (const float*)d_in[3];   // 1024 x 512
  const float* Wout = (const float*)d_in[4];   // 512 x 512
  const float* bout = (const float*)d_in[5];   // 512
  float* out = (float*)d_out;                  // 8 x 512 x 4096

  char* w = (char*)d_ws;                       // ~18 MiB used
  __bf16* ST    = (__bf16*)(w + 0);            //  4 MiB: S^T[b] 512x512 bf16
  __bf16* Tm    = (__bf16*)(w + 4194304);      //  4 MiB: T[b] = Wq*S, 512x512 bf16
  float*  dots  = (float*) (w + 8388608);      //  1 MiB: 64 x 64x64 f32
  __bf16* attnT = (__bf16*)(w + 9437184);      //  0.5 MiB
  __bf16* W2    = (__bf16*)(w + 9961472);      //  4 MiB
  __bf16* WvT   = (__bf16*)(w + 14155776);     //  0.5 MiB
  __bf16* W3    = (__bf16*)(w + 14680064);     //  4 MiB

  // 1) S^T[b] = (f_m[b] * f_n[b]^T)^T   K=4096
  wmma_gemm_nt<128,128,64,32,64, true,true,false, true,true,false>
    <<<dim3(4,4,8), 256, 0, stream>>>(
      f_m, 4096, 512LL*4096, 0,  f_n, 4096, 512LL*4096, 0,
      ST, 512, 512LL*512, 0,  4096, 1, 1.0f, nullptr);

  // 2) T[b] = Wq * S[b]  (= Wq * ST^T, NT)   K=512
  wmma_gemm_nt<128,128,64,32,64, true,false,false, true,false,false>
    <<<dim3(4,4,8), 256, 0, stream>>>(
      Wq, 512, 0, 0,  ST, 512, 512LL*512, 0,
      Tm, 512, 512LL*512, 0,  512, 1, 1.0f, nullptr);

  // 3) dots[b,h] = scale * T_h * Wk_h^T   (Wk = Wkv rows 0..511)   K=512
  wmma_gemm_nt<64,64,64,32,32, false,true,false, false,false,false>
    <<<dim3(1,1,64), 128, 0, stream>>>(
      Tm, 512, 512LL*512, 64LL*512,  Wkv, 512, 0, 64LL*512,
      dots, 64, 8LL*4096, 4096,  512, 8, 0.125f, nullptr);

  // 4) attn^T = softmax(dots) transposed, bf16
  softmax_transpose_64<<<64, 64, 0, stream>>>(dots, attnT);

  // Wv^T for step 6
  transpose_wv_bf16<<<1024, 256, 0, stream>>>(Wkv, WvT);

  // 5) W2[b][:, h*64+j] = Wout_h * attn_h   (B = attnT, NT)   K=64
  wmma_gemm_nt<64,64,64,32,32, true,false,false, true,false,false>
    <<<dim3(8,1,64), 128, 0, stream>>>(
      Wout, 512, 0, 64,  attnT, 64, 8LL*4096, 4096,
      W2, 512, 512LL*512, 64,  64, 8, 1.0f, nullptr);

  // 6) W3[b] = W2[b] * Wv   (B = WvT, NT)   K=512
  wmma_gemm_nt<128,128,64,32,64, false,false,false, true,false,false>
    <<<dim3(4,4,8), 256, 0, stream>>>(
      W2, 512, 512LL*512, 0,  WvT, 512, 0, 0,
      W3, 512, 512LL*512, 0,  512, 1, 1.0f, nullptr);

  // 7) out[b] = W3[b] * f_n[b] + bout   (NN via LDS transpose of f_n tiles)   K=512
  wmma_gemm_nt<128,128,64,32,64, false,true,true, false,false,true>
    <<<dim3(4,32,8), 256, 0, stream>>>(
      W3, 512, 512LL*512, 0,  f_n, 4096, 512LL*4096, 0,
      out, 4096, 512LL*4096, 0,  512, 1, 1.0f, bout);
}
